// CGCNN_36764920054171
// MI455X (gfx1250) — matrix operand order repaired and checked by reading, not compiled, not run on hardware
//
#include <hip/hip_runtime.h>
#include <hip/hip_bf16.h>

typedef unsigned short u16;
typedef unsigned int   u32;
typedef __attribute__((ext_vector_type(16))) __bf16 v16bf;
typedef __attribute__((ext_vector_type(8)))  float  v8f;

#define HID     256
#define NNODES  100000
#define NGRAPHS 64
#define LN_EPS  1e-5f

__device__ __forceinline__ u16 f2bf(float x) {
  u32 u = __float_as_uint(x);
  u32 r = (u + 0x7FFFu + ((u >> 16) & 1u)) >> 16;   // round-to-nearest-even
  return (u16)r;
}

union FragBF { v16bf v; uint4 q[2]; };

// Block-wide sum (works for blockDim 256 or 512). Contains barriers -> call uniformly.
__device__ __forceinline__ float blockReduceSum(float v, float* sbuf) {
  const int lane = threadIdx.x & 31;
  const int wid  = threadIdx.x >> 5;
#pragma unroll
  for (int o = 16; o >= 1; o >>= 1) v += __shfl_xor(v, o, 32);
  __syncthreads();                       // protect sbuf reuse across calls
  if (lane == 0) sbuf[wid] = v;
  __syncthreads();
  const int nw = (int)((blockDim.x + 31) >> 5);
  float r = ((int)threadIdx.x < nw) ? sbuf[threadIdx.x] : 0.0f;
  if (wid == 0) {
#pragma unroll
    for (int o = 16; o >= 1; o >>= 1) r += __shfl_xor(r, o, 32);
    if (lane == 0) sbuf[16] = r;
  }
  __syncthreads();
  return sbuf[16];
}

// WT[n*K + k] = bf16(W[k*N + n])  (transpose so B fragments are K-contiguous)
__global__ void transpose_bf16_kernel(const float* __restrict__ W, u16* __restrict__ WT,
                                      int K, int N) {
  int total = K * N;
  for (int i = blockIdx.x * blockDim.x + threadIdx.x; i < total;
       i += gridDim.x * blockDim.x) {
    int n = i / K, k = i - n * K;
    WT[i] = f2bf(W[k * N + n]);
  }
}

// h = relu(LN(x @ W + b));  one node per block (256 channels = 256 threads)
__launch_bounds__(256)
__global__ void node_embed_kernel(const float* __restrict__ x, const float* __restrict__ W,
                                  const float* __restrict__ b, const float* __restrict__ gma,
                                  const float* __restrict__ beta,
                                  float* __restrict__ h, u16* __restrict__ hb) {
  __shared__ float red[17];
  const int n = blockIdx.x, t = threadIdx.x;
  float acc = b[t];
#pragma unroll
  for (int k = 0; k < 5; ++k) acc += x[n * 5 + k] * W[k * HID + t];
  float s  = blockReduceSum(acc, red);
  float s2 = blockReduceSum(acc * acc, red);
  float mean = s * (1.0f / HID);
  float var  = s2 * (1.0f / HID) - mean * mean;
  float v = (acc - mean) * rsqrtf(var + LN_EPS) * gma[t] + beta[t];
  v = fmaxf(v, 0.0f);
  size_t idx = (size_t)n * HID + t;
  h[idx]  = v;
  hb[idx] = f2bf(v);
}

// Fused GEMM (bf16 WMMA, fp32 acc) + bias + LayerNorm (+ optional residual into h).
// Tile: M=32 rows x N=256 cols per workgroup; 8 waves, each computes a 2(M) x 2(N)
// grid of 16x16 WMMA tiles (A-frags reused across N, B-frags reused across M).
// A tile staged into LDS with CDNA5 async loads (ASYNCcnt-tracked, VGPR-bypassing).
// KDIM=256: out = LN(A0 @ WT^T + b)          -> outb (bf16)
// KDIM=512: v = LN([A0|A1] @ WT^T + b); h+=v -> hres (f32) and outb (bf16)
template <int KDIM, bool FUSE>
__launch_bounds__(256)
__global__ void gemm_ln_kernel(const u16* __restrict__ A0, const u16* __restrict__ A1,
                               const u16* __restrict__ WT, const float* __restrict__ bias,
                               const float* __restrict__ gma, const float* __restrict__ beta,
                               float* __restrict__ hres, u16* __restrict__ outb) {
  __shared__ __align__(16) u16 ldsA[32 * KDIM];
  __shared__ float ldsC[32 * HID];
  const int tid = threadIdx.x;
  const int m0  = blockIdx.x * 32;

  // Stage A tile (32 x KDIM bf16) into LDS via async global->LDS B128 copies.
  const int cpr = KDIM / 8;                 // 16B chunks per row
  const u32 ldsA_base = (u32)(size_t)(&ldsA[0]);
  for (int i = tid; i < 32 * cpr; i += 256) {
    const int row = i / cpr, kc = i - row * cpr;
    const u32 dst = ldsA_base + (u32)(row * KDIM + kc * 8) * 2u;
    if (KDIM == 512 && kc >= 32) {
      const u32 src = (u32)(((m0 + row) * HID + (kc - 32) * 8) * 2);
      asm volatile("global_load_async_to_lds_b128 %0, %1, %2"
                   :: "v"(dst), "v"(src), "s"(A1) : "memory");
    } else {
      const u32 src = (u32)(((m0 + row) * HID + kc * 8) * 2);
      asm volatile("global_load_async_to_lds_b128 %0, %1, %2"
                   :: "v"(dst), "v"(src), "s"(A0) : "memory");
    }
  }
  asm volatile("s_wait_asynccnt 0x0" ::: "memory");
  __syncthreads();

  const int wave = tid >> 5, lane = tid & 31;
  const int l15 = lane & 15, half = lane >> 4;
  const int colB0 = l15 + (wave * 2) * 16;
  const int colB1 = colB0 + 16;

  v8f acc00 = {}; v8f acc01 = {}; v8f acc10 = {}; v8f acc11 = {};
#pragma unroll
  for (int kb = 0; kb < KDIM; kb += 32) {
    FragBF a0, a1, b0, b1;
    // A frag: lane<16 holds K {kb+0..7, kb+16..23}; lane>=16 holds {kb+8..15, kb+24..31}
    const u16* ap0 = &ldsA[l15 * KDIM + kb + 8 * half];
    a0.q[0] = *(const uint4*)ap0;
    a0.q[1] = *(const uint4*)(ap0 + 16);
    const u16* ap1 = &ldsA[(l15 + 16) * KDIM + kb + 8 * half];
    a1.q[0] = *(const uint4*)ap1;
    a1.q[1] = *(const uint4*)(ap1 + 16);
    // B frag: lane = column, K {kb+8*half+0..7, kb+8*half+16..23} (same as A)
    const u16* bp0 = WT + (size_t)colB0 * KDIM + kb + 8 * half;
    b0.q[0] = *(const uint4*)bp0;
    b0.q[1] = *(const uint4*)(bp0 + 16);
    const u16* bp1 = WT + (size_t)colB1 * KDIM + kb + 8 * half;
    b1.q[0] = *(const uint4*)bp1;
    b1.q[1] = *(const uint4*)(bp1 + 16);
    acc00 = __builtin_amdgcn_wmma_f32_16x16x32_bf16(false, a0.v, false, b0.v, (short)0, acc00, false, false);
    acc01 = __builtin_amdgcn_wmma_f32_16x16x32_bf16(false, a0.v, false, b1.v, (short)0, acc01, false, false);
    acc10 = __builtin_amdgcn_wmma_f32_16x16x32_bf16(false, a1.v, false, b0.v, (short)0, acc10, false, false);
    acc11 = __builtin_amdgcn_wmma_f32_16x16x32_bf16(false, a1.v, false, b1.v, (short)0, acc11, false, false);
  }

  // C/D layout: VGPR r -> row r (lanes 0-15) or r+8 (lanes 16-31); col = lane&15 (+ tile off)
  const int rbase = half * 8;
#pragma unroll
  for (int r = 0; r < 8; ++r) {
    ldsC[(r + rbase) * HID + colB0]        = acc00[r] + bias[colB0];
    ldsC[(r + rbase) * HID + colB1]        = acc01[r] + bias[colB1];
    ldsC[(16 + r + rbase) * HID + colB0]   = acc10[r] + bias[colB0];
    ldsC[(16 + r + rbase) * HID + colB1]   = acc11[r] + bias[colB1];
  }
  __syncthreads();

  // LayerNorm: wave w handles rows 4w..4w+3; each lane covers 8 channels.
  for (int rr = 0; rr < 4; ++rr) {
    const int row = wave * 4 + rr;
    float s = 0.0f, s2 = 0.0f;
#pragma unroll
    for (int j = 0; j < 8; ++j) {
      float v = ldsC[row * HID + lane + 32 * j];
      s += v; s2 += v * v;
    }
#pragma unroll
    for (int o = 16; o >= 1; o >>= 1) {
      s  += __shfl_xor(s, o, 32);
      s2 += __shfl_xor(s2, o, 32);
    }
    const float mean = s * (1.0f / HID);
    const float var  = s2 * (1.0f / HID) - mean * mean;
    const float rs   = rsqrtf(var + LN_EPS);
#pragma unroll
    for (int j = 0; j < 8; ++j) {
      const int c = lane + 32 * j;
      float v = (ldsC[row * HID + c] - mean) * rs * gma[c] + beta[c];
      const size_t idx = (size_t)(m0 + row) * HID + c;
      if (FUSE) {
        float nh = hres[idx] + v;   // residual h += h_new
        hres[idx] = nh;
        outb[idx] = f2bf(nh);
      } else {
        outb[idx] = f2bf(v);        // ht (bf16 only; feeds next GEMM)
      }
    }
  }
}

// Segment-mean over sorted batch: one graph per block, binary-searched range.
__launch_bounds__(256)
__global__ void pool_kernel(const float* __restrict__ h, const int* __restrict__ batch,
                            float* __restrict__ pooled) {
  const int g = blockIdx.x, t = threadIdx.x;
  int lo = 0, hi = NNODES;
  while (lo < hi) { int mid = (lo + hi) >> 1; if (batch[mid] < g) lo = mid + 1; else hi = mid; }
  const int start = lo;
  hi = NNODES;
  while (lo < hi) { int mid = (lo + hi) >> 1; if (batch[mid] < g + 1) lo = mid + 1; else hi = mid; }
  const int end = lo;
  float acc = 0.0f;
  for (int n = start; n < end; ++n) acc += h[(size_t)n * HID + t];
  const float cnt = (float)(end - start);
  pooled[g * HID + t] = acc / fmaxf(cnt, 1.0f);
}

// add = L2( relu(LN(L1(additional_features))) )
__launch_bounds__(256)
__global__ void add_mlp_kernel(const float* __restrict__ af, const float* __restrict__ W1,
                               const float* __restrict__ b1, const float* __restrict__ gma,
                               const float* __restrict__ beta, const float* __restrict__ W2,
                               const float* __restrict__ b2, float* __restrict__ addout) {
  __shared__ float red[17];
  __shared__ float s1[HID];
  const int g = blockIdx.x, t = threadIdx.x;
  float acc = b1[t];
#pragma unroll
  for (int k = 0; k < 8; ++k) acc += af[g * 8 + k] * W1[k * HID + t];
  float s  = blockReduceSum(acc, red);
  float s2 = blockReduceSum(acc * acc, red);
  float mean = s * (1.0f / HID);
  float var  = s2 * (1.0f / HID) - mean * mean;
  float v = (acc - mean) * rsqrtf(var + LN_EPS) * gma[t] + beta[t];
  s1[t] = fmaxf(v, 0.0f);
  __syncthreads();
  float o = b2[t];
  for (int k = 0; k < HID; ++k) o += s1[k] * W2[k * HID + t];
  addout[g * HID + t] = o;
}

// z=[pooled|add]; z=relu(LN(z@W1+b1)); z=relu(z@W2+b2); out=z@W3+b3
__launch_bounds__(512)
__global__ void head_kernel(const float* __restrict__ pooled, const float* __restrict__ addf,
                            const float* __restrict__ W1, const float* __restrict__ b1,
                            const float* __restrict__ gma, const float* __restrict__ beta,
                            const float* __restrict__ W2, const float* __restrict__ b2,
                            const float* __restrict__ W3, const float* __restrict__ b3,
                            float* __restrict__ out) {
  __shared__ float red[17];
  __shared__ float z[2 * HID];
  __shared__ float z1s[2 * HID];
  __shared__ float z2s[HID];
  const int g = blockIdx.x, t = threadIdx.x;
  z[t] = (t < HID) ? pooled[g * HID + t] : addf[g * HID + (t - HID)];
  __syncthreads();
  float acc = b1[t];
  for (int k = 0; k < 2 * HID; ++k) acc += z[k] * W1[k * (2 * HID) + t];
  float s  = blockReduceSum(acc, red);
  float s2 = blockReduceSum(acc * acc, red);
  float mean = s * (1.0f / (2 * HID));
  float var  = s2 * (1.0f / (2 * HID)) - mean * mean;
  float v = (acc - mean) * rsqrtf(var + LN_EPS) * gma[t] + beta[t];
  z1s[t] = fmaxf(v, 0.0f);
  __syncthreads();
  if (t < HID) {
    float o = b2[t];
    for (int k = 0; k < 2 * HID; ++k) o += z1s[k] * W2[k * HID + t];
    z2s[t] = fmaxf(o, 0.0f);
  }
  __syncthreads();
  float part = (t < HID) ? z2s[t] * W3[t] : 0.0f;
  float tot = blockReduceSum(part, red);
  if (t == 0) out[g] = tot + b3[0];
}

// ---- Host ----------------------------------------------------------------
// d_in order (JAX pytree, dicts flattened in sorted-key order):
//   0:x 1:edge_index 2:edge_attr 3:batch 4:additional_features
//   5..10  : add_mlp.l1{W,b,be,g}, add_mlp.l2{W,b}
//   11+10i : convs[i]: edge{W,b}, node{W,b,be,g}, out{W,b,be,g}
//   41..44 : edge_emb{W,b,be,g}   (DEAD: edge path never reaches output)
//   45..52 : head.l1{W,b,be,g}, l2{W,b}, l3{W,b}
//   53..56 : node_emb{W,b,be,g}
extern "C" void kernel_launch(void* const* d_in, const int* in_sizes, int n_in,
                              void* d_out, int out_size, void* d_ws, size_t ws_size,
                              hipStream_t stream) {
  (void)in_sizes; (void)out_size; (void)ws_size;
  if (n_in < 57) return;

  const float* x     = (const float*)d_in[0];
  const int*   batch = (const int*)d_in[3];
  const float* addf  = (const float*)d_in[4];
  auto F = [&](int i) { return (const float*)d_in[i]; };

  char* ws = (char*)d_ws;
  size_t off = 0;
  auto carve = [&](size_t bytes) -> char* {
    char* p = ws + off;
    off += (bytes + 255) & ~(size_t)255;
    return p;
  };
  float* h    = (float*)carve((size_t)NNODES * HID * 4);   // 102.4 MB (L2-resident)
  u16*   hb   = (u16*)carve((size_t)NNODES * HID * 2);     // bf16 copy of h
  u16*   htb  = (u16*)carve((size_t)NNODES * HID * 2);     // bf16 ht
  u16*   WnT[3]; u16* WoT[3];
  for (int i = 0; i < 3; ++i) {
    WnT[i] = (u16*)carve((size_t)256 * 256 * 2);
    WoT[i] = (u16*)carve((size_t)512 * 256 * 2);
  }
  float* pooled = (float*)carve((size_t)NGRAPHS * HID * 4);
  float* addo   = (float*)carve((size_t)NGRAPHS * HID * 4);

  // Weight transpose + bf16 convert (node_mlp W: [256,256], output_mlp W: [512,256])
  for (int i = 0; i < 3; ++i) {
    const int base = 11 + 10 * i;
    transpose_bf16_kernel<<<256, 256, 0, stream>>>(F(base + 2), WnT[i], 256, 256);
    transpose_bf16_kernel<<<256, 256, 0, stream>>>(F(base + 6), WoT[i], 512, 256);
  }

  // Node embedding: Linear -> LN -> ReLU   (params: W=53 b=54 be=55 g=56)
  node_embed_kernel<<<NNODES, 256, 0, stream>>>(x, F(53), F(54), F(56), F(55), h, hb);

  // 3 conv layers; edge/gate/scatter path is dead code in the reference -> skipped.
  for (int i = 0; i < 3; ++i) {
    const int base = 11 + 10 * i;
    // ht = LN(h @ Wn + bn)        (node: W=+2 b=+3 be=+4 g=+5)
    gemm_ln_kernel<256, false><<<NNODES / 32, 256, 0, stream>>>(
        hb, nullptr, WnT[i], F(base + 3), F(base + 5), F(base + 4), nullptr, htb);
    // h += LN([h|ht] @ Wo + bo)   (out:  W=+6 b=+7 be=+8 g=+9)
    gemm_ln_kernel<512, true><<<NNODES / 32, 256, 0, stream>>>(
        hb, htb, WoT[i], F(base + 7), F(base + 9), F(base + 8), h, hb);
  }

  pool_kernel<<<NGRAPHS, 256, 0, stream>>>(h, batch, pooled);
  add_mlp_kernel<<<NGRAPHS, 256, 0, stream>>>(addf, F(5), F(6), F(8), F(7), F(9), F(10), addo);
  head_kernel<<<NGRAPHS, 512, 0, stream>>>(pooled, addo, F(45), F(46), F(48), F(47),
                                           F(49), F(50), F(51), F(52), (float*)d_out);
}
